// NodeInfoPropagator_10110353014861
// MI455X (gfx1250) — matrix operand order repaired and checked by reading, not compiled, not run on hardware
//
#include <hip/hip_runtime.h>
#include <hip/hip_bf16.h>
#include <math.h>

#define N_NODES 100000
#define DEPTH   2

typedef __attribute__((ext_vector_type(16))) _Float16 v16h;
typedef __attribute__((ext_vector_type(8)))  float    v8f;

// ---------------------------------------------------------------------------
// WMMA fragment loaders per CDNA5 ISA 7.12.2 (16-bit layouts, wave32)
// A 16x32: lanes 0-15 -> M=0..15, V0..3 hold K pairs {0,1}..{6,7};
//          lanes 16-31 same M with K+8; V4..7 add K+16.
__device__ __forceinline__ v16h load_fragA(const _Float16* tile, int ld,
                                           int lane, int k0) {
  union { v16h h; unsigned u[8]; } f;
  const int m = lane & 15, hf = lane >> 4;
  const _Float16* row = tile + m * ld + k0 + (hf << 3);
#pragma unroll
  for (int v = 0; v < 8; ++v) {
    const int k = ((v & 3) << 1) + ((v >> 2) << 4);
    f.u[v] = *(const unsigned*)(row + k);
  }
  return f.h;
}
// B 32x16: lanes 0-15 -> N=0..15, V0..3 hold K pairs {0,1}..{6,7};
//          lanes 16-31 same N with K+8; V4..7 add K+16.  wrow0 = row-major W tile (16 x ld),
//          B[k][n] = W[n][k].
__device__ __forceinline__ v16h load_fragB(const _Float16* wrow0, int ld,
                                           int lane, int k0) {
  union { v16h h; unsigned u[8]; } f;
  const int n = lane & 15, hf = lane >> 4;
  const _Float16* row = wrow0 + n * ld + k0 + (hf << 3);
#pragma unroll
  for (int v = 0; v < 8; ++v)
    f.u[v] = *(const unsigned*)(row + (((v & 3) << 1) + ((v >> 2) << 4)));
  return f.h;
}

// ---------------------------------------------------------------------------
// Convert the three fp32 weight matrices to f16 in workspace.
// layout: [0,16384) W_fc | [16384,65536) W_ih | [65536,114688) W_hh
__global__ __launch_bounds__(256) void cvt_kernel(const float* __restrict__ Wfc,
                                                  const float* __restrict__ Wih,
                                                  const float* __restrict__ Whh,
                                                  _Float16* __restrict__ dst) {
  const int i = blockIdx.x * 256 + threadIdx.x;
  if (i < 16384)       dst[i] = (_Float16)Wfc[i];
  else if (i < 65536)  dst[i] = (_Float16)Wih[i - 16384];
  else if (i < 114688) dst[i] = (_Float16)Whh[i - 65536];
}

// ---------------------------------------------------------------------------
// H = X @ W_fc^T + b_fc     (32 nodes / block, 8 waves)
__global__ __launch_bounds__(256) void fc_kernel(const float* __restrict__ X,
                                                 const _Float16* __restrict__ Wfc,
                                                 const float* __restrict__ bfc,
                                                 float* __restrict__ H) {
  __shared__ __align__(16) _Float16 At[32 * 128];
  const int tid  = threadIdx.x;
  const int base = blockIdx.x * 32;
#pragma unroll 4
  for (int i = tid; i < 32 * 128; i += 256)
    At[i] = (_Float16)X[(size_t)base * 128 + i];
  __syncthreads();

  const int wave = tid >> 5, lane = tid & 31;
  const int rt = wave >> 2;      // row tile 0..1 (16 nodes each)
  const int c0 = wave & 3;       // col tiles c0 and c0+4 (8 total)
  v16h a[4];
#pragma unroll
  for (int kc = 0; kc < 4; ++kc)
    a[kc] = load_fragA(At + rt * 16 * 128, 128, lane, kc * 32);

#pragma unroll
  for (int t = 0; t < 2; ++t) {
    const int ct = c0 + t * 4;
    v8f acc = {};
#pragma unroll
    for (int kc = 0; kc < 4; ++kc) {
      v16h b = load_fragB(Wfc + ct * 16 * 128, 128, lane, kc * 32);
      acc = __builtin_amdgcn_wmma_f32_16x16x32_f16(false, a[kc], false, b,
                                                   (short)0, acc, false, false);
    }
    const int col  = ct * 16 + (lane & 15);
    const float bb = bfc[col];
#pragma unroll
    for (int v = 0; v < 8; ++v) {
      const int row = base + rt * 16 + v + ((lane >> 4) << 3);
      H[(size_t)row * 128 + col] = acc[v] + bb;
    }
  }
}

// ---------------------------------------------------------------------------
// S[i] = relu(H)[parent_i] + mean over valid nbrs of relu(H)[nbr]
// 2 nodes / block, one lane per feature column; gathers hit L2 (H = 51 MB).
__global__ __launch_bounds__(256) void agg_kernel(const int* __restrict__ adj,
                                                  const float* __restrict__ H,
                                                  float* __restrict__ S) {
  const int node = blockIdx.x * 2 + (threadIdx.x >> 7);
  const int col  = threadIdx.x & 127;
  const int* row = adj + node * 17;
  const float p  = H[(size_t)row[0] * 128 + col];
  float acc = p > 0.f ? p : 0.f;
  float s = 0.f;
  int cnt = 0;
#pragma unroll
  for (int j = 1; j <= 16; ++j) {
    const int idx = row[j];
    if (idx >= 0) {
      const float v = H[(size_t)idx * 128 + col];
      s += (v > 0.f ? v : 0.f);
      ++cnt;
    }
  }
  S[(size_t)node * 128 + col] = acc + s / (float)cnt;   // deg >= 1 guaranteed
}

// ---------------------------------------------------------------------------
// Fused GRU step, in-place update of H (each block owns its 16 rows):
//   GI = H  @ W_ih^T   GH = S @ W_hh^T   (24 col tiles each, 8 waves x 6 tiles)
//   r = sig(gi_r+gh_r); z = sig(gi_z+gh_z); n = tanh(gi_n + r*gh_n)
//   H = (1-z)*n + z*S          (reference gates z against the summary)
__global__ __launch_bounds__(256) void gru_kernel(float* __restrict__ H,
                                                  const float* __restrict__ S,
                                                  const _Float16* __restrict__ Wih,
                                                  const _Float16* __restrict__ Whh,
                                                  const float* __restrict__ bih,
                                                  const float* __restrict__ bhh) {
  __shared__ __align__(16) _Float16 hA[16 * 128];
  __shared__ __align__(16) _Float16 sA[16 * 128];
  __shared__ float gbuf[2 * 16 * 384];        // [GI | GH] fp32 accum staging
  const int tid  = threadIdx.x;
  const int base = blockIdx.x * 16;

  for (int i = tid; i < 16 * 128; i += 256) {
    hA[i] = (_Float16)H[(size_t)base * 128 + i];
    sA[i] = (_Float16)S[(size_t)base * 128 + i];
  }
  __syncthreads();

  const int wave = tid >> 5, lane = tid & 31;
  v16h aH[4], aS[4];
#pragma unroll
  for (int kc = 0; kc < 4; ++kc) {
    aH[kc] = load_fragA(hA, 128, lane, kc * 32);
    aS[kc] = load_fragA(sA, 128, lane, kc * 32);
  }

#pragma unroll
  for (int i = 0; i < 6; ++i) {
    const int t   = wave + i * 8;        // 0..47
    const int mat = (t >= 24) ? 1 : 0;   // 0 = GI (uses H), 1 = GH (uses S)
    const int ct  = t - mat * 24;        // col tile 0..23
    const _Float16* W = (mat ? Whh : Wih) + ct * 16 * 128;
    v8f acc = {};
#pragma unroll
    for (int kc = 0; kc < 4; ++kc) {
      v16h b = load_fragB(W, 128, lane, kc * 32);
      acc = __builtin_amdgcn_wmma_f32_16x16x32_f16(false, mat ? aS[kc] : aH[kc],
                                                   false, b, (short)0, acc,
                                                   false, false);
    }
    float* g = gbuf + mat * (16 * 384);
    const int col = ct * 16 + (lane & 15);
#pragma unroll
    for (int v = 0; v < 8; ++v) {
      const int m = v + ((lane >> 4) << 3);
      g[m * 384 + col] = acc[v];
    }
  }
  __syncthreads();

  for (int e = tid; e < 16 * 128; e += 256) {
    const int m = e >> 7, d = e & 127;
    const float* gi = gbuf + m * 384;
    const float* gh = gbuf + 16 * 384 + m * 384;
    const float gir = gi[d]       + bih[d];
    const float giz = gi[128 + d] + bih[128 + d];
    const float gin = gi[256 + d] + bih[256 + d];
    const float ghr = gh[d]       + bhh[d];
    const float ghz = gh[128 + d] + bhh[128 + d];
    const float ghn = gh[256 + d] + bhh[256 + d];
    const float r = 1.f / (1.f + expf(-(gir + ghr)));
    const float z = 1.f / (1.f + expf(-(giz + ghz)));
    const float n = tanhf(gin + r * ghn);
    const size_t off = (size_t)(base + m) * 128 + d;
    H[off] = (1.f - z) * n + z * S[off];
  }
}

// ---------------------------------------------------------------------------
extern "C" void kernel_launch(void* const* d_in, const int* in_sizes, int n_in,
                              void* d_out, int out_size, void* d_ws, size_t ws_size,
                              hipStream_t stream) {
  const int*   adj = (const int*)  d_in[0];   // (N,17) int32
  const float* X   = (const float*)d_in[1];   // (N,128)
  const float* Wfc = (const float*)d_in[2];   // (128,128)
  const float* bfc = (const float*)d_in[3];   // (128,)
  const float* Wih = (const float*)d_in[4];   // (384,128)
  const float* Whh = (const float*)d_in[5];   // (384,128)
  const float* bih = (const float*)d_in[6];   // (384,)
  const float* bhh = (const float*)d_in[7];   // (384,)

  float* H = (float*)d_out;                   // h lives in d_out (N x 128 f32)
  float* S = (float*)d_ws;                    // summary: N*128*4 = 51.2 MB
  _Float16* W16 = (_Float16*)((char*)d_ws + (size_t)N_NODES * 128 * sizeof(float));
  _Float16* Wfc16 = W16;
  _Float16* Wih16 = W16 + 16384;
  _Float16* Whh16 = W16 + 65536;

  cvt_kernel<<<448, 256, 0, stream>>>(Wfc, Wih, Whh, W16);
  fc_kernel <<<N_NODES / 32, 256, 0, stream>>>(X, Wfc16, bfc, H);
  for (int d = 0; d < DEPTH; ++d) {
    agg_kernel<<<N_NODES / 2, 256, 0, stream>>>(adj, H, S);
    gru_kernel<<<N_NODES / 16, 256, 0, stream>>>(H, S, Wih16, Whh16, bih, bhh);
  }
}
